// ResolutionRegressionNet_63780264345755
// MI455X (gfx1250) — matrix operand order repaired and checked by reading, not compiled, not run on hardware
//
#include <hip/hip_runtime.h>
#include <hip/hip_bf16.h>
#include <math.h>

// ---------------- types ----------------
typedef _Float16 half_t;
typedef __attribute__((ext_vector_type(16))) _Float16 v16h;
typedef __attribute__((ext_vector_type(8)))  _Float16 v8h;
typedef __attribute__((ext_vector_type(2)))  _Float16 v2h;
typedef __attribute__((ext_vector_type(8)))  float    v8f;

#define BSZ   8
#define CH    128
#define HWPX  16384      // 128*128
#define HEADS 8
#define HIDP  352        // 340 padded to 352 (K mult of 32)
#define HID2P 704        // 680 padded to 704 (M mult of 32)

// ---------------- fragment helpers ----------------
// A-matrix 16x32 f16 per-lane: halves[0..15] = K {koff..koff+15};
// caller passes row ptr + koff.
__device__ inline v16h load_afrag(const half_t* p) {
  v8h lo = *(const v8h*)p;
  v8h hi = *(const v8h*)(p + 8);
  v16h a;
#pragma unroll
  for (int i = 0; i < 8; i++) { a[i] = lo[i]; a[i + 8] = hi[i]; }
  return a;
}
// 16 contiguous halves (B-matrix per-lane data / generic vector)
__device__ inline v16h load_hvec16(const half_t* p) {
  v8h lo = *(const v8h*)p;
  v8h hi = *(const v8h*)(p + 8);
  v16h r;
#pragma unroll
  for (int i = 0; i < 8; i++) { r[i] = lo[i]; r[i + 8] = hi[i]; }
  return r;
}

// ---------------- weight convert (with zero padding) ----------------
__global__ void cvt_pad_kernel(const float* __restrict__ src, half_t* __restrict__ dst,
                               int rs, int cs, int rd, int cd) {
  size_t i = (size_t)blockIdx.x * 256 + threadIdx.x;
  size_t tot = (size_t)rd * cd;
  if (i >= tot) return;
  int r = (int)(i / cd), c = (int)(i % cd);
  dst[i] = (r < rs && c < cs) ? (half_t)src[(size_t)r * cs + c] : (half_t)0.f;
}

// ---------- channel LayerNorm per pixel; outputs PIXEL-MAJOR f16 ----------
// input element (c,p) at x[b*CH*HWPX + c*cstr + p*pstr]
__global__ __launch_bounds__(256) void ln_cast_kernel(
    const float* __restrict__ x, const float* __restrict__ w, const float* __restrict__ bb,
    half_t* __restrict__ ln_out, half_t* __restrict__ raw_out, size_t cstr, size_t pstr) {
  int b = blockIdx.y;
  int p = blockIdx.x * 256 + threadIdx.x;
  const float* xp = x + (size_t)b * CH * HWPX + (size_t)p * pstr;
  float s = 0.f, s2 = 0.f;
#pragma unroll 4
  for (int c = 0; c < CH; c++) { float v = xp[(size_t)c * cstr]; s += v; s2 += v * v; }
  float mu = s * (1.f / CH);
  float var = s2 * (1.f / CH) - mu * mu;
  var = var < 0.f ? 0.f : var;
  float rstd = rsqrtf(var + 1e-5f);
  half_t* lnp = ln_out + ((size_t)b * HWPX + p) * CH;
  half_t* rwp = raw_out ? raw_out + ((size_t)b * HWPX + p) * CH : (half_t*)0;
  for (int c8 = 0; c8 < CH; c8 += 8) {
    v8h lh, rh;
#pragma unroll
    for (int i = 0; i < 8; i++) {
      float v = xp[(size_t)(c8 + i) * cstr];
      rh[i] = (half_t)v;
      lh[i] = (half_t)((v - mu) * rstd * w[c8 + i] + bb[c8 + i]);
    }
    *(v8h*)(lnp + c8) = lh;
    if (rwp) *(v8h*)(rwp + c8) = rh;
  }
}

// ---------------- WMMA GEMM (LDS-free, pixel-major activations) ------------
// Y[b][n][M] (modes 0..2) = Wt[M,K] * X[b][n][K]   (X pixel-major)
// mode 0: outh[(b*HW+n)*M + m]                  = acc            (f16 pm)
// mode 1: outh[(b*HW+n)*256 + off + m]          = acc + residCM  (f16 pm concat)
// mode 2: outf[(b*HW+n)*128 + m]                = acc            (f32 pm)
// mode 3: outf[(b*128+m)*HW + n] = nan_to_num(acc + residPM)     (f32 CHANNEL-major, final)
__global__ __launch_bounds__(256) void gemm_wmma_kernel(
    const half_t* __restrict__ Wt, const half_t* __restrict__ X, int M, int K,
    int mode, const float* __restrict__ resid, half_t* __restrict__ outh,
    float* __restrict__ outf, int out_ch_off) {
  int b = blockIdx.z;
  int nb = blockIdx.x * 256;
  int m_blk = blockIdx.y * 32;
  int tid = threadIdx.x, lane = tid & 31, wave = tid >> 5;
  int wm = wave & 1, wn = wave >> 1;            // 2 (M) x 4 (N) waves
  v8f acc[4] = {};
  const size_t bHW = (size_t)b * HWPX;
  int arow = m_blk + wm * 16 + (lane & 15);
  int koff = (lane >> 4) * 16;
  int kb   = (lane >> 4) * 16;
  const half_t* wrow = Wt + (size_t)arow * K;
  const half_t* xq[4];
#pragma unroll
  for (int j = 0; j < 4; j++) {
    int n = nb + wn * 64 + j * 16 + (lane & 15);
    xq[j] = X + (bHW + n) * K + kb;
  }
  for (int k0 = 0; k0 < K; k0 += 32) {
    v16h a = load_afrag(wrow + k0 + koff);
#pragma unroll
    for (int j = 0; j < 4; j++) {
      v16h bf = load_hvec16(xq[j] + k0);
      acc[j] = __builtin_amdgcn_wmma_f32_16x16x32_f16(false, a, false, bf,
                                                      (short)0, acc[j], false, false);
    }
  }
  // C/D layout: row = m_blk + wm*16 + r + 8*(lane>>4), col n = lane&15
  int rbase = m_blk + wm * 16 + 8 * (lane >> 4);
  int nbase = nb + wn * 64 + (lane & 15);
#pragma unroll
  for (int j = 0; j < 4; j++) {
    int n = nbase + j * 16;
    if (mode == 0) {
      v8h hv;
#pragma unroll
      for (int r = 0; r < 8; r++) hv[r] = (half_t)acc[j][r];
      *(v8h*)(outh + (bHW + n) * M + rbase) = hv;
    } else if (mode == 1) {
      v8h hv;
#pragma unroll
      for (int r = 0; r < 8; r++) {
        float v = acc[j][r] + resid[((size_t)b * CH + rbase + r) * HWPX + n];
        hv[r] = (half_t)v;
      }
      *(v8h*)(outh + (bHW + n) * 256 + out_ch_off + rbase) = hv;
    } else if (mode == 2) {
      *(v8f*)(outf + (bHW + n) * CH + rbase) = acc[j];
    } else {
      v8f rv = *(const v8f*)(resid + (bHW + n) * CH + rbase);
#pragma unroll
      for (int r = 0; r < 8; r++) {
        float v = acc[j][r] + rv[r];
        if (!isfinite(v)) v = 1e-5f;
        outf[((size_t)b * CH + rbase + r) * HWPX + n] = v;
      }
    }
  }
}

// -------- depthwise 3x3, pixel-major in/out: [b][n][Cch] ----------
__global__ __launch_bounds__(256) void dw3_kernel(
    const half_t* __restrict__ in, const float* __restrict__ w,
    half_t* __restrict__ out, int Cch) {
  int b = blockIdx.y;
  size_t idx = (size_t)blockIdx.x * 256 + threadIdx.x;
  int c = (int)(idx % Cch);
  int p = (int)(idx / Cch);
  int y = p >> 7, x = p & 127;
  const half_t* ip = in + (size_t)b * HWPX * Cch + c;
  const float* wp = w + c * 9;
  float acc = 0.f;
#pragma unroll
  for (int dy = -1; dy <= 1; dy++) {
    int yy = y + dy; if (yy < 0 || yy > 127) continue;
#pragma unroll
    for (int dx = -1; dx <= 1; dx++) {
      int xx = x + dx; if (xx < 0 || xx > 127) continue;
      acc += wp[(dy + 1) * 3 + (dx + 1)] * (float)ip[(size_t)(yy * 128 + xx) * Cch];
    }
  }
  out[((size_t)b * HWPX + p) * Cch + c] = (half_t)acc;
}

// ------- per-(b,head) Gram + norms + softmax; WMMA with per-wave LDS -------
// qd pixel-major [b][n][128]; kvd pixel-major [b][n][256] (k=0..127, v=128..255)
__global__ __launch_bounds__(256) void attn_stats_kernel(
    const half_t* __restrict__ qd, const half_t* __restrict__ kvd,
    const float* __restrict__ temp, float* __restrict__ attn_out) {
  __shared__ float G[256];
  __shared__ float Sq[16], Sk[16];
  __shared__ float Arow[256];
  __shared__ __attribute__((aligned(16))) half_t tr[8][2][16 * 40];  // per-wave q/k tiles [c][n32]
  int bh = blockIdx.x, b = bh >> 3, h = bh & 7;
  int tid = threadIdx.x, lane = tid & 31, wave = tid >> 5;
  G[tid] = 0.f;
  if (tid < 16) { Sq[tid] = 0.f; Sk[tid] = 0.f; }
  __syncthreads();

  const size_t bHW = (size_t)b * HWPX;
  half_t* lq = &tr[wave][0][0];
  half_t* lk = &tr[wave][1][0];
  v8f gqk = {}, gqq = {}, gkk = {};
  int row  = lane & 15;
  int koff = (lane >> 4) * 16;
  int kb   = (lane >> 4) * 16;
  for (int nc = wave * 2048; nc < wave * 2048 + 2048; nc += 32) {
    // per-wave transpose staging: 512 elems each, 8 half2 per lane per matrix
#pragma unroll
    for (int i = 0; i < 8; i++) {
      int e = i * 32 + lane;
      int nl = e >> 3, cp = (e & 7) * 2;
      v2h qv = *(const v2h*)(qd + (bHW + nc + nl) * 128 + h * 16 + cp);
      v2h kv = *(const v2h*)(kvd + (bHW + nc + nl) * 256 + h * 16 + cp);
      lq[cp * 40 + nl] = qv[0]; lq[(cp + 1) * 40 + nl] = qv[1];
      lk[cp * 40 + nl] = kv[0]; lk[(cp + 1) * 40 + nl] = kv[1];
    }
    asm volatile("s_wait_dscnt 0x0" ::: "memory");  // wave-private LDS fence
    v16h aq = load_afrag(lq + row * 40 + koff);
    v16h ak = load_afrag(lk + row * 40 + koff);
    v16h bq = load_hvec16(lq + row * 40 + kb);
    v16h bk = load_hvec16(lk + row * 40 + kb);
    gqk = __builtin_amdgcn_wmma_f32_16x16x32_f16(false, aq, false, bk, (short)0, gqk, false, false);
    gqq = __builtin_amdgcn_wmma_f32_16x16x32_f16(false, aq, false, bq, (short)0, gqq, false, false);
    gkk = __builtin_amdgcn_wmma_f32_16x16x32_f16(false, ak, false, bk, (short)0, gkk, false, false);
  }
#pragma unroll
  for (int r = 0; r < 8; r++) {
    int m = r + 8 * (lane >> 4), d = lane & 15;
    atomicAdd(&G[m * 16 + d], gqk[r]);
    if (m == d) { atomicAdd(&Sq[d], gqq[r]); atomicAdd(&Sk[d], gkk[r]); }
  }
  __syncthreads();
  {
    int c = tid >> 4, d = tid & 15;
    float nq = sqrtf(Sq[c]); nq = nq < 1e-12f ? 1e-12f : nq;
    float nk = sqrtf(Sk[d]); nk = nk < 1e-12f ? 1e-12f : nk;
    Arow[tid] = G[c * 16 + d] / (nq * nk) * temp[h];
  }
  __syncthreads();
  if (tid < 16) {
    float mx = -1e30f;
    for (int d = 0; d < 16; d++) mx = fmaxf(mx, Arow[tid * 16 + d]);
    float ssum = 0.f;
    for (int d = 0; d < 16; d++) { float e = __expf(Arow[tid * 16 + d] - mx); Arow[tid * 16 + d] = e; ssum += e; }
    float inv = 1.f / ssum;
    for (int d = 0; d < 16; d++)
      attn_out[(size_t)bh * 256 + tid * 16 + d] = Arow[tid * 16 + d] * inv;
  }
}

// ---- o = attn * v, pixel-major: per-thread pixel, all heads ----
__global__ __launch_bounds__(256) void attn_apply_kernel(
    const float* __restrict__ attn, const half_t* __restrict__ kvd,
    half_t* __restrict__ o) {
  __shared__ float A[HEADS * 256];
  int b = blockIdx.y;
  int tid = threadIdx.x;
#pragma unroll
  for (int i = 0; i < HEADS; i++)
    A[i * 256 + tid] = attn[((size_t)(b * HEADS + i)) * 256 + tid];
  __syncthreads();
  size_t n = (size_t)blockIdx.x * 256 + tid;
  const half_t* vp = kvd + ((size_t)b * HWPX + n) * 256 + 128;
  v16h vr[8];
#pragma unroll
  for (int i = 0; i < 8; i++) vr[i] = load_hvec16(vp + i * 16);
  half_t* op = o + ((size_t)b * HWPX + n) * 128;
#pragma unroll
  for (int hh = 0; hh < HEADS; hh++) {
    v8h o1, o2;
    v16h vh = vr[hh];
#pragma unroll
    for (int cc = 0; cc < 16; cc++) {
      const float* ar = &A[hh * 256 + cc * 16];
      float s = 0.f;
#pragma unroll
      for (int d = 0; d < 16; d++) s += ar[d] * (float)vh[d];
      if (cc < 8) o1[cc] = (half_t)s; else o2[cc - 8] = (half_t)s;
    }
    *(v8h*)(op + hh * 16) = o1;
    *(v8h*)(op + hh * 16 + 8) = o2;
  }
}

// ---------------- zero + mean pool (pixel-major y) ----------------
__global__ void zero_kernel(float* __restrict__ p, int n) {
  int i = blockIdx.x * 256 + threadIdx.x;
  if (i < n) p[i] = 0.f;
}
__global__ __launch_bounds__(256) void meanpool_kernel(const float* __restrict__ y,
                                                       float* __restrict__ ybar) {
  int b = blockIdx.y, chunk = blockIdx.x;   // 16 chunks of 1024 pixels
  int tid = threadIdx.x;
  int c = tid & 127, hh = tid >> 7;
  __shared__ float red[256];
  float s = 0.f;
  int n0 = chunk * 1024 + hh * 512;
  const float* yp = y + (size_t)b * HWPX * CH + c;
  for (int n = n0; n < n0 + 512; n++) s += yp[(size_t)n * CH];
  red[tid] = s;
  __syncthreads();
  if (tid < 128) atomicAdd(&ybar[b * CH + c], (red[tid] + red[tid + 128]) * (1.f / HWPX));
}

// ---------------- channel-attention MLP -> sigmoid scale ----------------
__global__ __launch_bounds__(128) void ca_kernel(
    const float* __restrict__ ybar, const float* __restrict__ w1, const float* __restrict__ b1,
    const float* __restrict__ w2, const float* __restrict__ b2, float* __restrict__ scale) {
  int b = blockIdx.x, tid = threadIdx.x;
  __shared__ float yv[128]; __shared__ float t1[8];
  yv[tid] = ybar[b * CH + tid];
  __syncthreads();
  if (tid < 8) {
    float s = b1[tid];
    for (int c = 0; c < CH; c++) s += w1[tid * CH + c] * yv[c];
    t1[tid] = s > 0.f ? s : 0.f;
  }
  __syncthreads();
  float s = b2[tid];
  for (int j = 0; j < 8; j++) s += w2[tid * 8 + j] * t1[j];
  scale[b * CH + tid] = 1.f / (1.f + __expf(-s));
}

// ---- x_out = y*scale + origin, y pixel-major in place; origin channel-major
__global__ __launch_bounds__(256) void scale_resid_kernel(
    float* __restrict__ y, const float* __restrict__ scale, const float* __restrict__ origin) {
  int b = blockIdx.y;
  int n = blockIdx.x * 256 + threadIdx.x;
  float* yp = y + ((size_t)b * HWPX + n) * CH;
#pragma unroll 4
  for (int c = 0; c < CH; c++)
    yp[c] = yp[c] * scale[b * CH + c] + origin[((size_t)b * CH + c) * HWPX + n];
}

// ---- fused FFN gate: dwconv3 + exact gelu gating, pixel-major h/g ----
__global__ __launch_bounds__(256) void ffn_gate_kernel(
    const half_t* __restrict__ hbuf, const float* __restrict__ dww, half_t* __restrict__ g) {
  int b = blockIdx.y;
  size_t idx = (size_t)blockIdx.x * 256 + threadIdx.x;   // over HWPX*HIDP
  int j = (int)(idx % HIDP);
  int p = (int)(idx / HIDP);
  half_t* gp = g + ((size_t)b * HWPX + p) * HIDP + j;
  if (j >= 340) { *gp = (half_t)0.f; return; }
  int y = p >> 7, x = p & 127;
  const half_t* h1 = hbuf + (size_t)b * HWPX * HID2P + j;
  const half_t* h2 = h1 + 340;
  const float* w1 = dww + j * 9;
  const float* w2 = dww + (340 + j) * 9;
  float a1 = 0.f, a2 = 0.f;
#pragma unroll
  for (int dy = -1; dy <= 1; dy++) {
    int yy = y + dy; if (yy < 0 || yy > 127) continue;
#pragma unroll
    for (int dx = -1; dx <= 1; dx++) {
      int xx = x + dx; if (xx < 0 || xx > 127) continue;
      size_t o = (size_t)(yy * 128 + xx) * HID2P;
      a1 += w1[(dy + 1) * 3 + (dx + 1)] * (float)h1[o];
      a2 += w2[(dy + 1) * 3 + (dx + 1)] * (float)h2[o];
    }
  }
  float ge = 0.5f * a1 * (1.f + erff(a1 * 0.70710678118f));   // exact gelu
  *gp = (half_t)(ge * a2);
}

// ================== host launcher ==================
extern "C" void kernel_launch(void* const* d_in, const int* in_sizes, int n_in,
                              void* d_out, int out_size, void* d_ws, size_t ws_size,
                              hipStream_t stream) {
  const float* origin  = (const float*)d_in[0];
  const float* low     = (const float*)d_in[1];
  const float* high    = (const float*)d_in[2];
  const float* norm1w  = (const float*)d_in[3];
  const float* norm1b  = (const float*)d_in[4];
  const float* norm_1w = (const float*)d_in[5];
  const float* norm_1b = (const float*)d_in[6];
  const float* norm2w  = (const float*)d_in[7];
  const float* norm2b  = (const float*)d_in[8];
  const float* a1_q    = (const float*)d_in[9];
  const float* a1_qdw  = (const float*)d_in[10];
  const float* a1_kv   = (const float*)d_in[11];
  const float* a1_kvdw = (const float*)d_in[12];
  const float* a1_po   = (const float*)d_in[13];
  const float* a1_temp = (const float*)d_in[14];
  const float* a2_q    = (const float*)d_in[15];
  const float* a2_qdw  = (const float*)d_in[16];
  const float* a2_kv   = (const float*)d_in[17];
  const float* a2_kvdw = (const float*)d_in[18];
  const float* a2_po   = (const float*)d_in[19];
  const float* a2_temp = (const float*)d_in[20];
  const float* conv1_w = (const float*)d_in[21];
  const float* ca1_w   = (const float*)d_in[22];
  const float* ca1_b   = (const float*)d_in[23];
  const float* ca2_w   = (const float*)d_in[24];
  const float* ca2_b   = (const float*)d_in[25];
  const float* ffn_pi  = (const float*)d_in[26];
  const float* ffn_dw  = (const float*)d_in[27];
  const float* ffn_po  = (const float*)d_in[28];
  (void)in_sizes; (void)n_in; (void)out_size; (void)ws_size;

  const size_t MB = 1ull << 20;
  char* ws = (char*)d_ws;
  // small region (< 4 MB)
  half_t* a1q_h   = (half_t*)(ws + 0);
  half_t* a1kv_h  = (half_t*)(ws + 32 * 1024);
  half_t* a1po_h  = (half_t*)(ws + 96 * 1024);
  half_t* a2q_h   = (half_t*)(ws + 128 * 1024);
  half_t* a2kv_h  = (half_t*)(ws + 160 * 1024);
  half_t* a2po_h  = (half_t*)(ws + 224 * 1024);
  half_t* conv1_h = (half_t*)(ws + 256 * 1024);
  half_t* ffnpi_h = (half_t*)(ws + 320 * 1024);   // 704*128*2 = 176 KB
  half_t* ffnpo_h = (half_t*)(ws + 512 * 1024);   // 128*352*2 = 88 KB
  float*  attn_s  = (float*)(ws + 640 * 1024);    // 8*8*256*4 = 64 KB
  float*  ybar    = (float*)(ws + 712 * 1024);
  float*  scalev  = (float*)(ws + 720 * 1024);
  // big tensors (lifetime-overlapped arena); all activations PIXEL-MAJOR
  half_t* low_h   = (half_t*)(ws + 4 * MB);    // 32 MB  [b][n][128]
  half_t* high_h  = (half_t*)(ws + 36 * MB);   // 32 MB
  half_t* ln_lo_h = (half_t*)(ws + 68 * MB);   // 32 MB
  half_t* ln_hi_h = (half_t*)(ws + 100 * MB);  // 32 MB
  half_t* qbuf    = (half_t*)(ws + 132 * MB);  // 32 MB
  half_t* kvbuf   = (half_t*)(ws + 164 * MB);  // 64 MB [b][n][256]
  half_t* qd      = (half_t*)(ws + 228 * MB);  // 32 MB
  half_t* kvd     = (half_t*)(ws + 260 * MB);  // 64 MB
  half_t* obuf    = (half_t*)(ws + 324 * MB);  // 32 MB
  half_t* xcat    = (half_t*)(ws + 356 * MB);  // 64 MB [b][n][256]
  float*  ybuf    = (float*)(ws + 420 * MB);   // 64 MB f32 [b][n][128] (y, then x_out)
  // phase E reuse (earlier buffers dead)
  half_t* ln2_h   = (half_t*)(ws + 4 * MB);    // 32 MB (over low_h)
  half_t* hbuf    = (half_t*)(ws + 36 * MB);   // 176 MB [b][n][704] (over high_h..kvbuf)
  half_t* gbuf    = (half_t*)(ws + 228 * MB);  // 88 MB [b][n][352] (over qd/kvd)
  float*  outp    = (float*)d_out;

  dim3 blk(256);

  // 0) weight converts (padded where needed)
  auto cvt = [&](const float* s, half_t* d, int rs, int cs, int rd, int cd) {
    int n = rd * cd;
    cvt_pad_kernel<<<(n + 255) / 256, blk, 0, stream>>>(s, d, rs, cs, rd, cd);
  };
  cvt(a1_q,  a1q_h,  128, 128, 128, 128);
  cvt(a1_kv, a1kv_h, 256, 128, 256, 128);
  cvt(a1_po, a1po_h, 128, 128, 128, 128);
  cvt(a2_q,  a2q_h,  128, 128, 128, 128);
  cvt(a2_kv, a2kv_h, 256, 128, 256, 128);
  cvt(a2_po, a2po_h, 128, 128, 128, 128);
  cvt(conv1_w, conv1_h, 128, 256, 128, 256);
  cvt(ffn_pi, ffnpi_h, 680, 128, HID2P, 128);
  cvt(ffn_po, ffnpo_h, 128, 340, 128, HIDP);

  // 1) LayerNorms + raw f16 casts (channel-major f32 in -> pixel-major f16 out)
  ln_cast_kernel<<<dim3(64, BSZ), blk, 0, stream>>>(low,  norm1w,  norm1b,  ln_lo_h, low_h,  HWPX, 1);
  ln_cast_kernel<<<dim3(64, BSZ), blk, 0, stream>>>(high, norm_1w, norm_1b, ln_hi_h, high_h, HWPX, 1);

  // ---- attn branch 1: q from high, kv from LN(low); resid low -> xcat[:,0:128]
  gemm_wmma_kernel<<<dim3(64, 4, BSZ), blk, 0, stream>>>(a1q_h,  high_h,  128, 128, 0, nullptr, qbuf,  nullptr, 0);
  gemm_wmma_kernel<<<dim3(64, 8, BSZ), blk, 0, stream>>>(a1kv_h, ln_lo_h, 256, 128, 0, nullptr, kvbuf, nullptr, 0);
  dw3_kernel<<<dim3(HWPX * 128 / 256, BSZ), blk, 0, stream>>>(qbuf,  a1_qdw,  qd,  128);
  dw3_kernel<<<dim3(HWPX * 256 / 256, BSZ), blk, 0, stream>>>(kvbuf, a1_kvdw, kvd, 256);
  attn_stats_kernel<<<BSZ * HEADS, blk, 0, stream>>>(qd, kvd, a1_temp, attn_s);
  attn_apply_kernel<<<dim3(64, BSZ), blk, 0, stream>>>(attn_s, kvd, obuf);
  gemm_wmma_kernel<<<dim3(64, 4, BSZ), blk, 0, stream>>>(a1po_h, obuf, 128, 128, 1, low, xcat, nullptr, 0);

  // ---- attn branch 2: q from low, kv from LN(high); resid high -> xcat[:,128:256]
  gemm_wmma_kernel<<<dim3(64, 4, BSZ), blk, 0, stream>>>(a2q_h,  low_h,   128, 128, 0, nullptr, qbuf,  nullptr, 0);
  gemm_wmma_kernel<<<dim3(64, 8, BSZ), blk, 0, stream>>>(a2kv_h, ln_hi_h, 256, 128, 0, nullptr, kvbuf, nullptr, 0);
  dw3_kernel<<<dim3(HWPX * 128 / 256, BSZ), blk, 0, stream>>>(qbuf,  a2_qdw,  qd,  128);
  dw3_kernel<<<dim3(HWPX * 256 / 256, BSZ), blk, 0, stream>>>(kvbuf, a2_kvdw, kvd, 256);
  attn_stats_kernel<<<BSZ * HEADS, blk, 0, stream>>>(qd, kvd, a2_temp, attn_s);
  attn_apply_kernel<<<dim3(64, BSZ), blk, 0, stream>>>(attn_s, kvd, obuf);
  gemm_wmma_kernel<<<dim3(64, 4, BSZ), blk, 0, stream>>>(a2po_h, obuf, 128, 128, 1, high, xcat, nullptr, 128);

  // ---- fuse: y = conv1(xcat); CA scale; x_out = y*s + origin (in place on y)
  gemm_wmma_kernel<<<dim3(64, 4, BSZ), blk, 0, stream>>>(conv1_h, xcat, 128, 256, 2, nullptr, nullptr, ybuf, 0);
  zero_kernel<<<4, blk, 0, stream>>>(ybar, BSZ * CH);
  meanpool_kernel<<<dim3(16, BSZ), blk, 0, stream>>>(ybuf, ybar);
  ca_kernel<<<BSZ, 128, 0, stream>>>(ybar, ca1_w, ca1_b, ca2_w, ca2_b, scalev);
  scale_resid_kernel<<<dim3(64, BSZ), blk, 0, stream>>>(ybuf, scalev, origin);

  // ---- FFN: LN -> pi-GEMM -> dw+gelu gate -> po-GEMM (+x_out, nan_to_num -> d_out)
  ln_cast_kernel<<<dim3(64, BSZ), blk, 0, stream>>>(ybuf, norm2w, norm2b, ln2_h, nullptr, 1, CH);
  gemm_wmma_kernel<<<dim3(64, HID2P / 32, BSZ), blk, 0, stream>>>(ffnpi_h, ln2_h, HID2P, 128, 0, nullptr, hbuf, nullptr, 0);
  ffn_gate_kernel<<<dim3(HWPX * HIDP / 256, BSZ), blk, 0, stream>>>(hbuf, ffn_dw, gbuf);
  gemm_wmma_kernel<<<dim3(64, 4, BSZ), blk, 0, stream>>>(ffnpo_h, gbuf, 128, HIDP, 3, ybuf, nullptr, outp, 0);
}
